// Attention_90615220011343
// MI455X (gfx1250) — matrix operand order repaired and checked by reading, not compiled, not run on hardware
//
#include <hip/hip_runtime.h>
#include <hip/hip_fp16.h>

typedef __attribute__((ext_vector_type(16))) _Float16 v16h;
typedef __attribute__((ext_vector_type(4)))  _Float16 v4h;
typedef __attribute__((ext_vector_type(8)))  float    v8f;

#define WMMA_F16(a,b,c) __builtin_amdgcn_wmma_f32_16x16x32_f16(false,(a),false,(b),(short)0,(c),false,false)

static constexpr int DHEAD = 32;
static constexpr int NPIX  = 4096;   // 64*64
static constexpr int CIN   = 256;
static constexpr int HID   = 128;    // 4 heads * 32
static constexpr int QKV3  = 384;

// ---------------------------------------------------------------------------
// Conversion kernels (run once per launch): x -> f16, weights -> f16 transposed
// so GEMM B-fragments become contiguous per-lane b128 loads.
// ---------------------------------------------------------------------------
__global__ __launch_bounds__(256) void k_cvt_x(const float* __restrict__ x,
                                               _Float16* __restrict__ xh) {
  int i = (blockIdx.x * 256 + threadIdx.x) * 4;    // 4096 blocks: 4,194,304 elems
  float4 v = *(const float4*)(x + i);
  v4h h;
  h[0] = (_Float16)v.x; h[1] = (_Float16)v.y;
  h[2] = (_Float16)v.z; h[3] = (_Float16)v.w;
  *(v4h*)(xh + i) = h;
}

__global__ __launch_bounds__(256) void k_cvt_w(const float* __restrict__ wq,
                                               const float* __restrict__ wo,
                                               _Float16* __restrict__ wqT,
                                               _Float16* __restrict__ woT) {
  int t = blockIdx.x * 256 + threadIdx.x;          // 512 blocks: 131072 threads
  if (t < CIN * QKV3) {                            // wqT[n][k] = wq[k][n]
    int k = t / QKV3, n = t % QKV3;
    wqT[(size_t)n * CIN + k] = (_Float16)wq[t];
  } else {                                         // woT[n][k] = wo[k][n]
    int u = t - CIN * QKV3;
    int k = u / CIN, n = u % CIN;
    woT[(size_t)n * HID + k] = (_Float16)wo[u];
  }
}

// ---------------------------------------------------------------------------
// Kernel 1: qkv = xh @ wqT^T (all f16), scatter into Q(scaled), K, V^T.
// Tile indices forced scalar via readfirstlane -> scalar epilogue branches.
// ---------------------------------------------------------------------------
__global__ __launch_bounds__(256) void k_qkv(const _Float16* __restrict__ xh,
                                             const _Float16* __restrict__ wqT,
                                             _Float16* __restrict__ Qs,
                                             _Float16* __restrict__ Kb,
                                             _Float16* __restrict__ Vt) {
  const int lane = threadIdx.x & 31;
  const int tile = __builtin_amdgcn_readfirstlane(blockIdx.x * 8 + (threadIdx.x >> 5));
  const int mt = tile / 12, nt = tile % 12;        // 1024 x 12 tiles
  const int m0 = mt * 16, n0 = nt * 32;            // SGPRs
  const int hi = lane >> 4, ln = lane & 15;

  v8f c0{}, c1{};
  const _Float16* xr = xh  + (size_t)(m0 + ln) * CIN;
  const _Float16* w0 = wqT + (size_t)(n0 + ln) * CIN;        // column n0+ln
  const _Float16* w1 = wqT + (size_t)(n0 + 16 + ln) * CIN;   // column n0+16+ln
  for (int k0 = 0; k0 < CIN; k0 += 32) {
    v16h a, b0, b1;
#pragma unroll
    for (int i = 0; i < 16; ++i)                   // A: contiguous 16 halves
      a[i] = xr[k0 + 16 * hi + i];
#pragma unroll
    for (int i = 0; i < 16; ++i) {                 // B: contiguous 16 halves
      b0[i] = w0[k0 + 16 * hi + i];
      b1[i] = w1[k0 + 16 * hi + i];
    }
    c0 = WMMA_F16(a, b0, c0);
    c1 = WMMA_F16(a, b1, c1);
  }

  const float scale = 0.17677669529663687f;        // 32^-0.5 folded into Q
#pragma unroll
  for (int t = 0; t < 2; ++t) {
    v8f c = t ? c1 : c0;
    const int base   = n0 + 16 * t;                // scalar
    const int region = base >> 7;                  // 0=Q 1=K 2=V (scalar branch)
    const int hh     = (base >> 5) & 3;            // head (scalar)
    const int d      = (base & 31) + ln;           // lane-varying d
    if (region == 0) {
#pragma unroll
      for (int r = 0; r < 8; ++r) {
        int m = m0 + r + 8 * hi, b = m >> 12, pix = m & 4095;
        Qs[(((size_t)(b * 4 + hh)) * NPIX + pix) * DHEAD + d] = (_Float16)(c[r] * scale);
      }
    } else if (region == 1) {
#pragma unroll
      for (int r = 0; r < 8; ++r) {
        int m = m0 + r + 8 * hi, b = m >> 12, pix = m & 4095;
        Kb[(((size_t)(b * 4 + hh)) * NPIX + pix) * DHEAD + d] = (_Float16)c[r];
      }
    } else {
#pragma unroll
      for (int r = 0; r < 8; ++r) {
        int m = m0 + r + 8 * hi, b = m >> 12, pix = m & 4095;
        Vt[(((size_t)(b * 4 + hh)) * DHEAD + d) * NPIX + pix] = (_Float16)c[r];
      }
    }
  }
}

// ---------------------------------------------------------------------------
// Kernel 2: flash attention, double-buffered GLOBAL_LOAD_ASYNC_TO_LDS_B128
// staging (ASYNCcnt) overlapping the WMMA+softmax of the current chunk.
// S computed transposed (K-tile @ Q^T) so exp'd accumulators feed the P@V
// WMMA A-operand in-place.
// ---------------------------------------------------------------------------
#define KST 40   // LDS row stride (halves) for K chunk, conflict-free
#define VST 72   // LDS row stride (halves) for V^T chunk, conflict-free

__global__ __launch_bounds__(256) void k_attn(const _Float16* __restrict__ Qs,
                                              const _Float16* __restrict__ Kb,
                                              const _Float16* __restrict__ Vt,
                                              _Float16* __restrict__ Ob) {
  __shared__ _Float16 Ks[2][64 * KST];
  __shared__ _Float16 Vs[2][32 * VST];
  const int lane = threadIdx.x & 31;
  const int hi = lane >> 4, ln = lane & 15;
  const int bh = blockIdx.y;                       // 0..15 = (batch, head)
  const int b = bh >> 2, h = bh & 3;
  const int q0 = __builtin_amdgcn_readfirstlane((blockIdx.x * 8 + (threadIdx.x >> 5)) * 16);

  const _Float16* Qg = Qs + ((size_t)bh * NPIX + q0) * DHEAD;
  const _Float16* Kg = Kb + (size_t)bh * NPIX * DHEAD;
  const _Float16* Vg = Vt + (size_t)bh * DHEAD * NPIX;

  // per-thread staging slots (fixed for the whole loop)
  const int srow = threadIdx.x >> 2, scg = threadIdx.x & 3;   // K: 64x32 halves
  const int sd   = threadIdx.x >> 3, scg2 = threadIdx.x & 7;  // Vt: 32x64 halves
  const unsigned ksl[2] = {(unsigned)(uintptr_t)&Ks[0][srow * KST + 8 * scg],
                           (unsigned)(uintptr_t)&Ks[1][srow * KST + 8 * scg]};
  const unsigned vsl[2] = {(unsigned)(uintptr_t)&Vs[0][sd * VST + 8 * scg2],
                           (unsigned)(uintptr_t)&Vs[1][sd * VST + 8 * scg2]};
  const _Float16* kg_base = Kg + (size_t)srow * DHEAD + 8 * scg;
  const _Float16* vg_base = Vg + (size_t)sd * NPIX + 8 * scg2;

  v16h bq;                                          // B-frag = Q^T (held all loop)
#pragma unroll
  for (int i = 0; i < 16; ++i) bq[i] = Qg[(size_t)ln * DHEAD + 16 * hi + i];

  v8f o0{}, o1{};
  float m_run = -1e30f, l_run = 0.f;

  // preload chunk 0 into buffer 0
  asm volatile("global_load_async_to_lds_b128 %0, %1, off"
               :: "v"(ksl[0]), "v"(kg_base) : "memory");
  asm volatile("global_load_async_to_lds_b128 %0, %1, off"
               :: "v"(vsl[0]), "v"(vg_base) : "memory");

  int buf = 0;
  for (int kk0 = 0; kk0 < NPIX; kk0 += 64, buf ^= 1) {
    asm volatile("s_wait_asynccnt 0x0" ::: "memory");  // my slice of cur is in LDS
    __syncthreads();                                   // everyone's slice is in;
                                                       // everyone done with 'next' buf
    if (kk0 + 64 < NPIX) {                             // prefetch next chunk async
      const _Float16* kg = kg_base + (size_t)(kk0 + 64) * DHEAD;
      const _Float16* vg = vg_base + (kk0 + 64);
      asm volatile("global_load_async_to_lds_b128 %0, %1, off"
                   :: "v"(ksl[buf ^ 1]), "v"(kg) : "memory");
      asm volatile("global_load_async_to_lds_b128 %0, %1, off"
                   :: "v"(vsl[buf ^ 1]), "v"(vg) : "memory");
    }
    const _Float16* Kc = &Ks[buf][0];
    const _Float16* Vc = &Vs[buf][0];

    v8f s[4];                                       // S^T tiles: 4 x 16 keys
#pragma unroll
    for (int t = 0; t < 4; ++t) {
      v16h a;
      const _Float16* kr = Kc + (16 * t + ln) * KST;
#pragma unroll
      for (int i = 0; i < 16; ++i) a[i] = kr[16 * hi + i];
      v8f z{};
      s[t] = WMMA_F16(a, bq, z);
    }

    // ---- online softmax (lane holds query ln; half-waves split the keys) ----
    float cm = -1e30f;
#pragma unroll
    for (int t = 0; t < 4; ++t)
#pragma unroll
      for (int r = 0; r < 8; ++r) cm = fmaxf(cm, s[t][r]);
    cm = fmaxf(cm, __shfl_xor(cm, 16, 32));
    float m_new = fmaxf(m_run, cm);
    float resc = __expf(m_run - m_new);
    float rs = 0.f;
#pragma unroll
    for (int t = 0; t < 4; ++t)
#pragma unroll
      for (int r = 0; r < 8; ++r) {
        float p = __expf(s[t][r] - m_new);
        s[t][r] = p;
        rs += p;
      }
    rs += __shfl_xor(rs, 16, 32);
    l_run = l_run * resc + rs;
    m_run = m_new;
#pragma unroll
    for (int r = 0; r < 8; ++r) {                   // rescale O (rows are regs)
      float f = __shfl(resc, r + 8 * hi, 32);
      o0[r] *= f;
      o1[r] *= f;
    }

    // ---- O += P @ V : P frags come straight from the S^T accumulators ----
#pragma unroll
    for (int g = 0; g < 2; ++g) {
      v16h pf, bv0, bv1;
#pragma unroll
      for (int i = 0; i < 8; ++i) {
        pf[i]     = (_Float16)s[2 * g][i];
        pf[i + 8] = (_Float16)s[2 * g + 1][i];
      }
      const _Float16* v0 = Vc + ln * VST + 32 * g + 8 * hi;
      const _Float16* v1 = Vc + (16 + ln) * VST + 32 * g + 8 * hi;
#pragma unroll
      for (int i = 0; i < 16; ++i) { bv0[i] = v0[((i >> 3) << 4) + (i & 7)]; bv1[i] = v1[((i >> 3) << 4) + (i & 7)]; }
      o0 = WMMA_F16(pf, bv0, o0);
      o1 = WMMA_F16(pf, bv1, o1);
    }
  }

  float linv = 1.f / l_run;
#pragma unroll
  for (int r = 0; r < 8; ++r) {
    float f = __shfl(linv, r + 8 * hi, 32);
    int q = q0 + r + 8 * hi;
    _Float16* orow = Ob + ((size_t)b * NPIX + q) * HID + h * DHEAD;
    orow[ln]      = (_Float16)(o0[r] * f);
    orow[16 + ln] = (_Float16)(o1[r] * f);
  }
}

// ---------------------------------------------------------------------------
// Kernel 3: out = Ob @ woT^T + b_out  (f16 -> f32)
// ---------------------------------------------------------------------------
__global__ __launch_bounds__(256) void k_out(const _Float16* __restrict__ Ob,
                                             const _Float16* __restrict__ woT,
                                             const float* __restrict__ bias,
                                             float* __restrict__ out) {
  const int lane = threadIdx.x & 31;
  const int tile = __builtin_amdgcn_readfirstlane(blockIdx.x * 8 + (threadIdx.x >> 5));
  const int mt = tile >> 3, nt = tile & 7;         // 1024 x 8 tiles
  const int m0 = mt * 16, n0 = nt * 32;
  const int hi = lane >> 4, ln = lane & 15;

  v8f c0{}, c1{};
  const _Float16* ar = Ob  + (size_t)(m0 + ln) * HID;
  const _Float16* w0 = woT + (size_t)(n0 + ln) * HID;
  const _Float16* w1 = woT + (size_t)(n0 + 16 + ln) * HID;
  for (int k0 = 0; k0 < HID; k0 += 32) {
    v16h a, b0, b1;
#pragma unroll
    for (int i = 0; i < 16; ++i) a[i] = ar[k0 + 16 * hi + i];
#pragma unroll
    for (int i = 0; i < 16; ++i) {
      b0[i] = w0[k0 + 16 * hi + i];
      b1[i] = w1[k0 + 16 * hi + i];
    }
    c0 = WMMA_F16(a, b0, c0);
    c1 = WMMA_F16(a, b1, c1);
  }
  float bb0 = bias[n0 + ln], bb1 = bias[n0 + 16 + ln];
#pragma unroll
  for (int r = 0; r < 8; ++r) {
    int m = m0 + r + 8 * hi;
    out[(size_t)m * CIN + n0 + ln]      = c0[r] + bb0;
    out[(size_t)m * CIN + n0 + 16 + ln] = c1[r] + bb1;
  }
}

// ---------------------------------------------------------------------------
extern "C" void kernel_launch(void* const* d_in, const int* in_sizes, int n_in,
                              void* d_out, int out_size, void* d_ws, size_t ws_size,
                              hipStream_t stream) {
  const float* x     = (const float*)d_in[0];
  const float* w_qkv = (const float*)d_in[1];
  const float* w_out = (const float*)d_in[2];
  const float* b_out = (const float*)d_in[3];
  float* out = (float*)d_out;

  char* ws = (char*)d_ws;
  _Float16* Qs  = (_Float16*)(ws);                                 // 4 MiB
  _Float16* Kb  = (_Float16*)(ws + (size_t)4  * 1024 * 1024);      // 4 MiB
  _Float16* Vt  = (_Float16*)(ws + (size_t)8  * 1024 * 1024);      // 4 MiB
  _Float16* Ob  = (_Float16*)(ws + (size_t)12 * 1024 * 1024);      // 4 MiB
  _Float16* xh  = (_Float16*)(ws + (size_t)16 * 1024 * 1024);      // 8 MiB
  _Float16* wqT = (_Float16*)(ws + (size_t)24 * 1024 * 1024);      // 192 KiB
  _Float16* woT = (_Float16*)(ws + (size_t)24 * 1024 * 1024 + 512 * 1024); // 64 KiB

  k_cvt_x<<<4096, 256, 0, stream>>>(x, xh);
  k_cvt_w<<<512, 256, 0, stream>>>(w_qkv, w_out, wqT, woT);
  k_qkv<<<1536, 256, 0, stream>>>(xh, wqT, Qs, Kb, Vt);
  k_attn<<<dim3(32, 16), 256, 0, stream>>>(Qs, Kb, Vt, Ob);
  k_out<<<1024, 256, 0, stream>>>(Ob, woT, b_out, out);
}
